// Encoder_43301860278639
// MI455X (gfx1250) — compile-verified, matrix operand order checked by reading, not run on hardware
//
#include <hip/hip_runtime.h>

// ---------------------------------------------------------------------------
// CGConv x2 + BatchNorm + global mean pool for MI455X (gfx1250, wave32, WMMA)
//
// N=50000 nodes, E=800000 edges, D=128, DE=64, Zin=320, G=256 graphs.
// GEMM work: 262 GFLOP total -> v_wmma_f32_16x16x32_f16 (f32 accumulate).
// Memory floor ~1.7GB/layer -> gather/scatter kernels stream at bandwidth.
// ---------------------------------------------------------------------------

#define D    128
#define DE   64
#define ZIN  (2*D + DE)      // 320
#define KT   (ZIN / 32)      // 10 K-tiles of 32
#define NT   (D / 16)        // 8 N-tiles of 16
#define FRAG_ELEMS (KT * NT * 512)   // halves per swizzled weight matrix = 40960
#define EPSV 1e-5f

typedef __attribute__((ext_vector_type(16))) _Float16 v16h;
typedef __attribute__((ext_vector_type(8)))  _Float16 v8h;
typedef __attribute__((ext_vector_type(4)))  _Float16 v4h;
typedef __attribute__((ext_vector_type(8)))  float    v8f;

// ---------------------------------------------------------------------------
// Swizzle W[Zin][D] (row-major f32) into per-lane WMMA B-fragments (f16).
// Fragment layout matches 16-bit A/B VGPR layout from CDNA5 ISA 7.12.2:
//   k = kt*32 + (h&7) + ((h>>3)<<4) + ((lane>>4)<<3),  n = nt*16 + (lane&15)
// Stored so a wave reads one fragment as 32 contiguous bytes per lane.
// ---------------------------------------------------------------------------
__global__ void prep_frag(const float* __restrict__ W, _Float16* __restrict__ out) {
    int idx = blockIdx.x * blockDim.x + threadIdx.x;
    if (idx >= FRAG_ELEMS) return;
    int h    = idx & 15;
    int lane = (idx >> 4) & 31;
    int ktnt = idx >> 9;
    int nt   = ktnt & 7;
    int kt   = ktnt >> 3;
    int k = kt * 32 + (h & 7) + ((h >> 3) << 4) + ((lane >> 4) << 3);
    int n = nt * 16 + (lane & 15);
    out[idx] = (_Float16)W[k * D + n];
}

__global__ void zero_f32(float* __restrict__ p, long n) {
    long i = (long)blockIdx.x * blockDim.x + threadIdx.x;
    long step = (long)gridDim.x * blockDim.x;
    for (; i < n; i += step) p[i] = 0.f;
}

// ---------------------------------------------------------------------------
// Edge message kernel: 512 threads = 16 waves.
//   wave w: matrix = w>>3 (0 = lin_f, 1 = lin_s), n-tile = w&7.
// Per 16-edge tile: float4 gather of z=[x_dst|x_src|ea] to LDS (f32->f16),
// preload all 10 A fragments (one dscnt wait), 10 back-to-back WMMAs with
// register-resident B, results staged through LDS, fused fast-math
// sigmoid*softplus epilogue, atomic f32 scatter into agg[dst].
// ---------------------------------------------------------------------------
__global__ __launch_bounds__(512) void edge_kernel(
    const float* __restrict__ x,
    const int*   __restrict__ srcI, const int* __restrict__ dstI,
    const float* __restrict__ ea,
    const _Float16* __restrict__ fragF, const _Float16* __restrict__ fragS,
    const float* __restrict__ bf, const float* __restrict__ bs,
    float* __restrict__ agg, int E, int ntiles)
{
    __shared__ __align__(32) _Float16 zt[16][ZIN + 8];  // pad 8 -> conflict-free ds_load_b128
    __shared__ float msg[2][16][D];                     // raw GEMM outputs (16 KB)

    const int tid  = threadIdx.x;
    const int lane = tid & 31;
    const int wv   = tid >> 5;       // 0..15
    const int mat  = wv >> 3;        // 0: f-gate, 1: s-core
    const int nt   = wv & 7;
    const _Float16* frag = mat ? fragS : fragF;

    // B fragments: resident in VGPRs for the whole kernel (10 x v16h = 80 VGPRs)
    v16h B[KT];
#pragma unroll
    for (int kt = 0; kt < KT; ++kt)
        B[kt] = *(const v16h*)(frag + (kt * NT + nt) * 512 + lane * 16);

    const int mrow = lane & 15;
    const int koff = (lane >> 4) << 3;

    for (int t = blockIdx.x; t < ntiles; t += gridDim.x) {
        const int e0 = t << 4;

        // ---- gather 16 x 320 f32 -> f16 LDS tile, float4 granules
        // (all segment boundaries are multiples of 4 columns)
        for (int i = tid; i < 16 * (ZIN / 4); i += 512) {
            int r = i / (ZIN / 4), c = (i - r * (ZIN / 4)) * 4;
            int e = e0 + r;
            float4 v = make_float4(0.f, 0.f, 0.f, 0.f);
            if (e < E) {
                if (c < D)          v = *(const float4*)&x[dstI[e] * D + c];
                else if (c < 2 * D) v = *(const float4*)&x[srcI[e] * D + (c - D)];
                else                v = *(const float4*)&ea[e * DE + (c - 2 * D)];
            }
            v4h h4 = { (_Float16)v.x, (_Float16)v.y, (_Float16)v.z, (_Float16)v.w };
            *(v4h*)&zt[r][c] = h4;
        }
        __syncthreads();

        // ---- preload all A fragments (20 ds_load_b128, single wait), then
        //      10 back-to-back v_wmma_f32_16x16x32_f16 on the accumulator chain
        v16h a[KT];
#pragma unroll
        for (int kt = 0; kt < KT; ++kt) {
            const _Float16* rp = &zt[mrow][kt * 32 + koff];
            ((v8h*)&a[kt])[0] = *(const v8h*)rp;          // K = kb+koff   .. +7
            ((v8h*)&a[kt])[1] = *(const v8h*)(rp + 16);   // K = kb+koff+16 .. +23
        }
        v8f acc = {};
#pragma unroll
        for (int kt = 0; kt < KT; ++kt)
            acc = __builtin_amdgcn_wmma_f32_16x16x32_f16(
                false, a[kt], false, B[kt], (short)0, acc, false, false);

        // ---- stage raw accumulators: VGPR i, lanes0-15 -> M=i, lanes16-31 -> M=i+8
        {
            int n = nt * 16 + (lane & 15);
            int mb = (lane >> 4) << 3;
#pragma unroll
            for (int i = 0; i < 8; ++i) msg[mat][mb + i][n] = acc[i];
        }
        __syncthreads();   // also guards zt reuse next iteration

        // ---- fused epilogue: sigmoid(f+bf) * softplus(s+bs), scatter-add
        for (int i = tid; i < 16 * D; i += 512) {
            int m = i >> 7, n = i & (D - 1);
            int e = e0 + m;
            if (e < E) {
                float f = msg[0][m][n] + bf[n];
                float s = msg[1][m][n] + bs[n];
                float sig = __builtin_amdgcn_rcpf(1.f + __expf(-f));
                float sp  = (s > 20.f) ? s : __logf(1.f + __expf(s));
                atomicAdd(&agg[dstI[e] * D + n], sig * sp);
            }
        }
    }
}

// ---------------------------------------------------------------------------
// BN stats: v = x + agg (residual), per-channel sum / sumsq, write v to vout.
// One channel per thread (block reads contiguous 512B rows), rows
// grid-strided; 2 atomics per thread total.
// ---------------------------------------------------------------------------
__global__ __launch_bounds__(D) void stats_kernel(
    const float* __restrict__ xin, const float* __restrict__ agg,
    float* __restrict__ vout, float* __restrict__ stats, int Nn)
{
    int ch = threadIdx.x;
    float s = 0.f, s2 = 0.f;
    for (int r = blockIdx.x; r < Nn; r += gridDim.x) {
        float v = xin[r * D + ch] + agg[r * D + ch];
        vout[r * D + ch] = v;
        s += v; s2 += v * v;
    }
    atomicAdd(&stats[ch], s);
    atomicAdd(&stats[D + ch], s2);
}

__global__ void bn_finalize(const float* __restrict__ stats,
                            const float* __restrict__ g, const float* __restrict__ be,
                            float* __restrict__ ss, int Nn)
{
    int ch = threadIdx.x;
    if (ch < D) {
        float inv = 1.f / (float)Nn;
        float mu  = stats[ch] * inv;
        float var = stats[D + ch] * inv - mu * mu;
        float a   = rsqrtf(var + EPSV) * g[ch];
        ss[ch]     = a;
        ss[D + ch] = be[ch] - mu * a;
    }
}

__global__ void bn_apply(float* __restrict__ v, const float* __restrict__ ss, long total) {
    long i = (long)blockIdx.x * blockDim.x + threadIdx.x;
    long step = (long)gridDim.x * blockDim.x;
    for (; i < total; i += step) {
        int ch = (int)(i & (D - 1));
        v[i] = v[i] * ss[ch] + ss[D + ch];
    }
}

// ---------------------------------------------------------------------------
// Global mean pool
// ---------------------------------------------------------------------------
__global__ __launch_bounds__(D) void pool_acc(
    const float* __restrict__ v, const int* __restrict__ batch,
    float* __restrict__ out, float* __restrict__ cnt, int Nn)
{
    for (int r = blockIdx.x; r < Nn; r += gridDim.x) {
        int g = batch[r];
        atomicAdd(&out[g * D + threadIdx.x], v[r * D + threadIdx.x]);
        if (threadIdx.x == 0) atomicAdd(&cnt[g], 1.f);
    }
}

__global__ void pool_div(float* __restrict__ out, const float* __restrict__ cnt, int total) {
    int i = blockIdx.x * blockDim.x + threadIdx.x;
    if (i < total) out[i] /= fmaxf(cnt[i >> 7], 1.f);
}

// ---------------------------------------------------------------------------
extern "C" void kernel_launch(void* const* d_in, const int* in_sizes, int n_in,
                              void* d_out, int out_size, void* d_ws, size_t ws_size,
                              hipStream_t stream) {
    const float* x     = (const float*)d_in[0];
    const int*   ei    = (const int*)d_in[1];
    const float* ea    = (const float*)d_in[2];
    const int*   batch = (const int*)d_in[3];
    const float* Wf0 = (const float*)d_in[4],  *bf0 = (const float*)d_in[5];
    const float* Ws0 = (const float*)d_in[6],  *bs0 = (const float*)d_in[7];
    const float* g0  = (const float*)d_in[8],  *be0 = (const float*)d_in[9];
    const float* Wf1 = (const float*)d_in[10], *bf1 = (const float*)d_in[11];
    const float* Ws1 = (const float*)d_in[12], *bs1 = (const float*)d_in[13];
    const float* g1  = (const float*)d_in[14], *be1 = (const float*)d_in[15];

    const int Nn = in_sizes[0] / D;
    const int E  = in_sizes[1] / 2;
    const int* srcI = ei;
    const int* dstI = ei + E;
    float* out = (float*)d_out;

    // workspace carve-up
    char* ws = (char*)d_ws;
    size_t off = 0;
    float* agg   = (float*)(ws + off); off += (size_t)Nn * D * 4;
    float* xbuf  = (float*)(ws + off); off += (size_t)Nn * D * 4;
    float* stats = (float*)(ws + off); off += 2 * D * 4;
    float* ssbuf = (float*)(ws + off); off += 2 * D * 4;
    float* cnt   = (float*)(ws + off); off += 1024;
    _Float16* fF0 = (_Float16*)(ws + off); off += FRAG_ELEMS * 2;
    _Float16* fS0 = (_Float16*)(ws + off); off += FRAG_ELEMS * 2;
    _Float16* fF1 = (_Float16*)(ws + off); off += FRAG_ELEMS * 2;
    _Float16* fS1 = (_Float16*)(ws + off); off += FRAG_ELEMS * 2;

    const int ntiles = (E + 15) / 16;
    const long ND = (long)Nn * D;
    dim3 b256(256), b512(512), b128(D);
    int fragBlocks = (FRAG_ELEMS + 255) / 256;

    // weight swizzle (once per launch; tiny)
    prep_frag<<<fragBlocks, b256, 0, stream>>>(Wf0, fF0);
    prep_frag<<<fragBlocks, b256, 0, stream>>>(Ws0, fS0);
    prep_frag<<<fragBlocks, b256, 0, stream>>>(Wf1, fF1);
    prep_frag<<<fragBlocks, b256, 0, stream>>>(Ws1, fS1);

    // ---- layer 0
    zero_f32<<<8192, b256, 0, stream>>>(agg, ND);
    zero_f32<<<1, b256, 0, stream>>>(stats, 2 * D);
    edge_kernel<<<4096, b512, 0, stream>>>(x, srcI, dstI, ea, fF0, fS0, bf0, bs0,
                                           agg, E, ntiles);
    stats_kernel<<<2048, b128, 0, stream>>>(x, agg, xbuf, stats, Nn);
    bn_finalize<<<1, b128, 0, stream>>>(stats, g0, be0, ssbuf, Nn);
    bn_apply<<<8192, b256, 0, stream>>>(xbuf, ssbuf, ND);

    // ---- layer 1
    zero_f32<<<8192, b256, 0, stream>>>(agg, ND);
    zero_f32<<<1, b256, 0, stream>>>(stats, 2 * D);
    edge_kernel<<<4096, b512, 0, stream>>>(xbuf, srcI, dstI, ea, fF1, fS1, bf1, bs1,
                                           agg, E, ntiles);
    stats_kernel<<<2048, b128, 0, stream>>>(xbuf, agg, xbuf, stats, Nn);
    bn_finalize<<<1, b128, 0, stream>>>(stats, g1, be1, ssbuf, Nn);
    bn_apply<<<8192, b256, 0, stream>>>(xbuf, ssbuf, ND);

    // ---- global mean pool
    zero_f32<<<(out_size + 255) / 256, b256, 0, stream>>>(out, out_size);
    zero_f32<<<1, b256, 0, stream>>>(cnt, 256);
    pool_acc<<<8192, b128, 0, stream>>>(xbuf, batch, out, cnt, Nn);
    pool_div<<<(out_size + 255) / 256, b256, 0, stream>>>(out, cnt, out_size);
}
